// LSTMCellCustom_70411693851113
// MI455X (gfx1250) — matrix operand order repaired and checked by reading, not compiled, not run on hardware
//
#include <hip/hip_runtime.h>

typedef __attribute__((ext_vector_type(16))) __bf16 v16bf;
typedef __attribute__((ext_vector_type(8)))  float  v8f;
typedef __attribute__((ext_vector_type(8)))  unsigned short u16x8;
typedef __attribute__((ext_vector_type(16))) unsigned short u16x16;
typedef __attribute__((ext_vector_type(4)))  unsigned short u16x4;
typedef __attribute__((ext_vector_type(4)))  int v4i;

#define BATCH 8192
#define HID   1024
#define KTOT  2048

#define GLOBAL_AS __attribute__((address_space(1)))
#define LDS_AS    __attribute__((address_space(3)))

#ifndef __has_builtin
#define __has_builtin(x) 0
#endif
#if __has_builtin(__builtin_amdgcn_global_load_async_to_lds_b128)
#define HAVE_ASYNC_LDS 1
#else
#define HAVE_ASYNC_LDS 0
#endif

__device__ __forceinline__ unsigned short f2bf(float f) {
  unsigned int u = __builtin_bit_cast(unsigned int, f);
  u += 0x7FFFu + ((u >> 16) & 1u);   // round-to-nearest-even
  return (unsigned short)(u >> 16);
}

__device__ __forceinline__ float sigm(float x) {
  return 1.0f / (1.0f + __expf(-x));
}
__device__ __forceinline__ float tanh_f(float x) {
  float e = __expf(-2.0f * fabsf(x));
  float t = (1.0f - e) / (1.0f + e);
  return copysignf(t, x);
}

__device__ __forceinline__ v16bf make_frag(u16x8 lo, u16x8 hi) {
  u16x16 t = __builtin_shufflevector(lo, hi, 0,1,2,3,4,5,6,7,8,9,10,11,12,13,14,15);
  return __builtin_bit_cast(v16bf, t);
}

// ---------------------------------------------------------------------------
// Pre-pass A: XHb[8192][2048] bf16 = concat(x, h) row-major.
// ---------------------------------------------------------------------------
__global__ __launch_bounds__(256)
void cvt_xh_kernel(const float* __restrict__ x, const float* __restrict__ h,
                   unsigned short* __restrict__ XHb) {
  const int gid  = blockIdx.x * 256 + threadIdx.x;
  const int flat = gid * 16;                 // 16 elements per thread
  const int row  = flat >> 11;
  const int col  = flat & 2047;
  const float* src = (col < 1024) ? (x + (size_t)row * 1024 + col)
                                  : (h + (size_t)row * 1024 + (col - 1024));
  float4 f0 = *reinterpret_cast<const float4*>(src);
  float4 f1 = *reinterpret_cast<const float4*>(src + 4);
  float4 f2 = *reinterpret_cast<const float4*>(src + 8);
  float4 f3 = *reinterpret_cast<const float4*>(src + 12);
  u16x8 a, b;
  a[0]=f2bf(f0.x); a[1]=f2bf(f0.y); a[2]=f2bf(f0.z); a[3]=f2bf(f0.w);
  a[4]=f2bf(f1.x); a[5]=f2bf(f1.y); a[6]=f2bf(f1.z); a[7]=f2bf(f1.w);
  b[0]=f2bf(f2.x); b[1]=f2bf(f2.y); b[2]=f2bf(f2.z); b[3]=f2bf(f2.w);
  b[4]=f2bf(f3.x); b[5]=f2bf(f3.y); b[6]=f2bf(f3.z); b[7]=f2bf(f3.w);
  *reinterpret_cast<u16x8*>(XHb + flat)     = a;
  *reinterpret_cast<u16x8*>(XHb + flat + 8) = b;
}

// ---------------------------------------------------------------------------
// Pre-pass B: WbT[3][1024][2048] bf16, WbT[g][n][k] = W{x|h}gate[k%1024][n].
// LDS-tiled 32x32 transpose+convert; 6 source matrices.
// ---------------------------------------------------------------------------
__global__ __launch_bounds__(256)
void cvt_wT_kernel(const float* __restrict__ Wxi, const float* __restrict__ Whi,
                   const float* __restrict__ Wxc, const float* __restrict__ Whc,
                   const float* __restrict__ Wxo, const float* __restrict__ Who,
                   unsigned short* __restrict__ WbT) {
  __shared__ unsigned short tile[32][36];

  const int m  = blockIdx.x >> 10;           // 0..5 source matrix
  const int t  = blockIdx.x & 1023;
  const int k0 = (t >> 5) * 32;
  const int n0 = (t & 31) * 32;
  const int g    = m >> 1;
  const int half = m & 1;                    // 0 = x-part (k 0..1023), 1 = h-part

  const float* src = (m == 0) ? Wxi : (m == 1) ? Whi : (m == 2) ? Wxc
                   : (m == 3) ? Whc : (m == 4) ? Wxo : Who;

  const int tid = threadIdx.x;
  {
    int kl = tid >> 3;
    int n4 = (tid & 7) * 4;
    float4 v = *reinterpret_cast<const float4*>(src + (size_t)(k0 + kl) * 1024 + n0 + n4);
    tile[kl][n4 + 0] = f2bf(v.x);
    tile[kl][n4 + 1] = f2bf(v.y);
    tile[kl][n4 + 2] = f2bf(v.z);
    tile[kl][n4 + 3] = f2bf(v.w);
  }
  __syncthreads();
  {
    int nl = tid >> 3;
    int k4 = (tid & 7) * 4;
    u16x4 p = { tile[k4 + 0][nl], tile[k4 + 1][nl], tile[k4 + 2][nl], tile[k4 + 3][nl] };
    size_t off = (size_t)(g * 1024 + n0 + nl) * 2048 + half * 1024 + k0 + k4;
    *reinterpret_cast<u16x4*>(WbT + off) = p;
  }
}

// ---------------------------------------------------------------------------
// Kernel 0: Z2[8192,8] = relu(relu(XH @ W1 + b1) @ W2 + b2)  (WMMA over K=2048)
// ---------------------------------------------------------------------------
__global__ __launch_bounds__(256)
void mlp_head_kernel(const unsigned short* __restrict__ XHb,
                     const float* __restrict__ W1, const float* __restrict__ b1,
                     const float* __restrict__ W2, const float* __restrict__ b2,
                     float* __restrict__ Z2) {
  __shared__ unsigned short W1T[16 * 2048];   // [j][k] bf16
  __shared__ float z1s[128][16];

  const int tid  = threadIdx.x;
  const int lane = tid & 31;
  const int wave = tid >> 5;
  const int rowBase = blockIdx.x * 128;

  // Stage W1 (2048x16 f32 row-major) -> LDS transposed bf16 [16][2048]
#pragma unroll 4
  for (int i = 0; i < 32; ++i) {
    int idx4 = tid + i * 256;
    int flat = idx4 * 4;
    int k  = flat >> 4;
    int j0 = flat & 15;
    float4 v = *reinterpret_cast<const float4*>(W1 + flat);
    W1T[(j0 + 0) * 2048 + k] = f2bf(v.x);
    W1T[(j0 + 1) * 2048 + k] = f2bf(v.y);
    W1T[(j0 + 2) * 2048 + k] = f2bf(v.z);
    W1T[(j0 + 3) * 2048 + k] = f2bf(v.w);
  }
  __syncthreads();

  const int m16    = lane & 15;
  const int hiHalf = lane >> 4;
  const int row    = rowBase + wave * 16 + m16;

  v8f acc = {};
  for (int kk = 0; kk < 64; ++kk) {
    const unsigned short* ap = XHb + (size_t)row * 2048 + kk * 32 + hiHalf * 8;
    u16x8 alo = *reinterpret_cast<const u16x8*>(ap);
    u16x8 ahi = *reinterpret_cast<const u16x8*>(ap + 16);
    v16bf av = make_frag(alo, ahi);

    const unsigned short* bp = &W1T[m16 * 2048 + kk * 32 + hiHalf * 8];
    u16x8 blo = *reinterpret_cast<const u16x8*>(bp);
    u16x8 bhi = *reinterpret_cast<const u16x8*>(bp + 16);
    v16bf bv = make_frag(blo, bhi);

    acc = __builtin_amdgcn_wmma_f32_16x16x32_bf16(false, av, false, bv,
                                                  (short)0, acc, false, false);
  }

  const float b1v = b1[m16];
#pragma unroll
  for (int r = 0; r < 8; ++r) {
    int mrow = wave * 16 + r + hiHalf * 8;
    z1s[mrow][m16] = fmaxf(acc[r] + b1v, 0.0f);
  }
  __syncthreads();

  {
    int rowL = tid >> 1;
    int jj0  = (tid & 1) * 4;
    float out[4];
#pragma unroll
    for (int u = 0; u < 4; ++u) {
      int jj = jj0 + u;
      float s = b2[jj];
#pragma unroll
      for (int j = 0; j < 16; ++j) s += z1s[rowL][j] * W2[j * 8 + jj];
      out[u] = fmaxf(s, 0.0f);
    }
    *reinterpret_cast<float4*>(Z2 + (size_t)(rowBase + rowL) * 8 + jj0) =
        make_float4(out[0], out[1], out[2], out[3]);
  }
}

// ---------------------------------------------------------------------------
// Kernel 1: fused 3-gate GEMM [8192,2048]@[2048,3x1024] + LSTM cell update.
// Block = 256 rows x 32 cols x 3 gates; 8 waves, each a 32x32 slab per gate.
// A tiles staged via async global->LDS copies; B frags straight from WbT.
// ---------------------------------------------------------------------------
__global__ __launch_bounds__(256)
void lstm_gates_kernel(const unsigned short* __restrict__ XHb,
                       const unsigned short* __restrict__ WbT,
                       const float* __restrict__ c,
                       const float* __restrict__ bi, const float* __restrict__ bc,
                       const float* __restrict__ bo,
                       const float* __restrict__ W3, const float* __restrict__ b3,
                       const float* __restrict__ Z2,
                       float* __restrict__ outH, float* __restrict__ outC,
                       float* __restrict__ outF) {
  __shared__ unsigned short Abuf[256 * 32];    // [row][k] bf16, 16 KB
  __shared__ float z2s[256][8];                // 8 KB

  const int tid  = threadIdx.x;
  const int lane = tid & 31;
  const int wave = tid >> 5;
  const int rowBase = blockIdx.x * 256;
  const int nBase   = blockIdx.y * 32;

  // Stage Z2 for this row block (256x8 f32)
#pragma unroll
  for (int i = 0; i < 2; ++i) {
    int flat4 = (tid + i * 256) * 4;
    int r = flat4 >> 3;
    int o = flat4 & 7;
    *reinterpret_cast<float4*>(&z2s[r][o]) =
        *reinterpret_cast<const float4*>(Z2 + (size_t)(rowBase + r) * 8 + o);
  }

  const int m16    = lane & 15;
  const int hiHalf = lane >> 4;

  v8f acc[3][2][2] = {};   // [gate][ni][mi]

  for (int kk = 0; kk < 64; ++kk) {
    __syncthreads();   // previous iteration's frag reads complete before overwrite

    // ---- stage A tile: 256 rows x 32 k bf16 (16 KB) ----
#pragma unroll
    for (int i = 0; i < 4; ++i) {
      int idx = tid + i * 256;            // 0..1023 16-byte chunks
      int r   = idx >> 2;
      int c8  = (idx & 3) * 8;
      const unsigned short* gsrc = XHb + (size_t)(rowBase + r) * 2048 + kk * 32 + c8;
      unsigned short* ldst = &Abuf[r * 32 + c8];
#if HAVE_ASYNC_LDS
      __builtin_amdgcn_global_load_async_to_lds_b128(
          (GLOBAL_AS v4i*)gsrc, (LDS_AS v4i*)ldst, 0, 0);
#else
      *reinterpret_cast<u16x8*>(ldst) = *reinterpret_cast<const u16x8*>(gsrc);
#endif
      if (i == 0) __builtin_prefetch(gsrc + 32, 0, 3);   // next K tile
    }
#if HAVE_ASYNC_LDS
#if __has_builtin(__builtin_amdgcn_s_wait_asynccnt)
    __builtin_amdgcn_s_wait_asynccnt(0);
#else
    asm volatile("s_wait_asynccnt 0" ::: "memory");
#endif
#endif
    __syncthreads();

    // ---- A fragments: this wave's 32 rows ----
    v16bf av[2];
#pragma unroll
    for (int mi = 0; mi < 2; ++mi) {
      const unsigned short* ap = &Abuf[(wave * 32 + mi * 16 + m16) * 32 + hiHalf * 8];
      u16x8 alo = *reinterpret_cast<const u16x8*>(ap);        // K = off .. off+7
      u16x8 ahi = *reinterpret_cast<const u16x8*>(ap + 16);   // K = off+16 .. off+23
      av[mi] = make_frag(alo, ahi);
    }

    // ---- B fragments straight from pre-transposed bf16 weights ----
#pragma unroll
    for (int g = 0; g < 3; ++g) {
#pragma unroll
      for (int ni = 0; ni < 2; ++ni) {
        const unsigned short* bp =
            WbT + (size_t)(g * 1024 + nBase + ni * 16 + m16) * 2048 + kk * 32 + hiHalf * 8;
        u16x8 blo = *reinterpret_cast<const u16x8*>(bp);
        u16x8 bhi = *reinterpret_cast<const u16x8*>(bp + 16);
        v16bf bv = make_frag(blo, bhi);
        if (g == 0 && ni == 0) __builtin_prefetch(bp + 32, 0, 3);
#pragma unroll
        for (int mi = 0; mi < 2; ++mi) {
          acc[g][ni][mi] = __builtin_amdgcn_wmma_f32_16x16x32_bf16(
              false, av[mi], false, bv, (short)0, acc[g][ni][mi], false, false);
        }
      }
    }
  }

  // ---- fused epilogue: f = sigmoid(Z2 @ W3 + b3), then cell update ----
#pragma unroll
  for (int ni = 0; ni < 2; ++ni) {
    int n = nBase + ni * 16 + m16;
    float bi_v = bi[n], bc_v = bc[n], bo_v = bo[n], b3_v = b3[n];
    float w3c[8];
#pragma unroll
    for (int j = 0; j < 8; ++j) w3c[j] = W3[j * 1024 + n];
#pragma unroll
    for (int mi = 0; mi < 2; ++mi) {
#pragma unroll
      for (int r = 0; r < 8; ++r) {
        int mrow = wave * 32 + mi * 16 + r + hiHalf * 8;
        int row  = rowBase + mrow;
        float fp = b3_v;
#pragma unroll
        for (int j = 0; j < 8; ++j) fp += z2s[mrow][j] * w3c[j];
        float fv = sigm(fp);
        float iv = sigm(acc[0][ni][mi][r] + bi_v);
        float gv = tanh_f(acc[1][ni][mi][r] + bc_v);
        float ov = sigm(acc[2][ni][mi][r] + bo_v);
        float cv = c[(size_t)row * 1024 + n];
        float cn = fv * cv + iv * gv;
        float hn = ov * tanh_f(cn);
        size_t oidx = (size_t)row * 1024 + n;
        outH[oidx] = hn;
        outC[oidx] = cn;
        outF[oidx] = fv;
      }
    }
  }
}

extern "C" void kernel_launch(void* const* d_in, const int* in_sizes, int n_in,
                              void* d_out, int out_size, void* d_ws, size_t ws_size,
                              hipStream_t stream) {
  const float* x   = (const float*)d_in[0];
  const float* h   = (const float*)d_in[1];
  const float* c   = (const float*)d_in[2];
  const float* Whi = (const float*)d_in[3];
  const float* Wxi = (const float*)d_in[4];
  const float* bi  = (const float*)d_in[5];
  const float* Whc = (const float*)d_in[6];
  const float* Wxc = (const float*)d_in[7];
  const float* bc  = (const float*)d_in[8];
  const float* Who = (const float*)d_in[9];
  const float* Wxo = (const float*)d_in[10];
  const float* bo  = (const float*)d_in[11];
  const float* W1  = (const float*)d_in[12];
  const float* b1  = (const float*)d_in[13];
  const float* W2  = (const float*)d_in[14];
  const float* b2  = (const float*)d_in[15];
  const float* W3  = (const float*)d_in[16];
  const float* b3  = (const float*)d_in[17];

  float* outH = (float*)d_out;
  float* outC = outH + (size_t)BATCH * HID;
  float* outF = outC + (size_t)BATCH * HID;

  // workspace layout
  char* ws = (char*)d_ws;
  unsigned short* XHb = (unsigned short*)ws;                               // 32 MB
  unsigned short* WbT = (unsigned short*)(ws + (size_t)32 * 1024 * 1024);  // 12 MB
  float* Z2 = (float*)(ws + (size_t)44 * 1024 * 1024 + 65536);             // 256 KB

  cvt_xh_kernel<<<dim3(4096), dim3(256), 0, stream>>>(x, h, XHb);
  cvt_wT_kernel<<<dim3(6144), dim3(256), 0, stream>>>(Wxi, Whi, Wxc, Whc, Wxo, Who, WbT);
  mlp_head_kernel<<<dim3(64), dim3(256), 0, stream>>>(XHb, W1, b1, W2, b2, Z2);
  lstm_gates_kernel<<<dim3(32, 32), dim3(256), 0, stream>>>(
      XHb, WbT, c, bi, bc, bo, W3, b3, Z2, outH, outC, outF);
}
